// GNN_35304631173261
// MI455X (gfx1250) — matrix operand order repaired and checked by reading, not compiled, not run on hardware
//
#include <hip/hip_runtime.h>
#include <hip/hip_bf16.h>
#include <math.h>

typedef _Float16 v16h __attribute__((ext_vector_type(16)));
typedef _Float16 v8h  __attribute__((ext_vector_type(8)));
typedef _Float16 v4h  __attribute__((ext_vector_type(4)));
typedef float    v8f  __attribute__((ext_vector_type(8)));

#define NHID   128
#define HEADS  8
#define DKD    16
#define TTYPES 3
#define RREL   4
#define MAXLEN 240
#define RTED   256   // 2*NHID

__device__ __forceinline__ float4 ld4(const float* p) { return *(const float4*)p; }

// ---------------------------------------------------------------------------
// A-fragment loader (CDNA5 16-bit A 16x32 layout, wave32):
// lane L: row = L&15; holds K = kb..kb+7 and 16+kb..16+kb+7, kb = (L>>4)*8
// ---------------------------------------------------------------------------
__device__ __forceinline__ v16h ldsA_frag(const _Float16* sA, int lda, int kc) {
  int lane = threadIdx.x & 31;
  int row  = lane & 15;
  int kb   = (lane >> 4) << 3;
  const _Float16* p = sA + row * lda + kc * 32 + kb;
  v8h lo = *(const v8h*)(p);
  v8h hi = *(const v8h*)(p + 16);
  v16h a;
#pragma unroll
  for (int i = 0; i < 8; ++i) { a[i] = lo[i]; a[i + 8] = hi[i]; }
  return a;
}

// ---------------------------------------------------------------------------
// Convert f32 weights [T,K,128] -> f16 fragment-major layout so each lane can
// load its WMMA B-fragment (16 contiguous halves) directly from global/L2.
// frag element: out[(((t*nk+kc)*8+ct)*32+lane)*16+i]
//   = W[t][kc*32 + (lane>>4)*8 + (i&7) + (i>>3)*16][ct*16 + (lane&15)]
// (B 32x16 layout: lane L -> col = L&15, K = kb..kb+7, 16+kb..16+kb+7, kb = (L>>4)*8)
// ---------------------------------------------------------------------------
__global__ void convert_w(const float* __restrict__ W, _Float16* __restrict__ out,
                          int K, int total)
{
  int gid = blockIdx.x * blockDim.x + threadIdx.x;
  if (gid >= total) return;
  int nk   = K >> 5;
  int i    = gid & 15;
  int lane = (gid >> 4) & 31;
  int ct   = (gid >> 9) & 7;
  int rest = gid >> 12;
  int kc   = rest % nk;
  int t    = rest / nk;
  int k    = kc * 32 + ((lane >> 4) << 3) + (i & 7) + ((i >> 3) << 4);
  int col  = (ct << 4) + (lane & 15);
  out[gid] = (_Float16)W[((size_t)t * K + k) * NHID + col];
}

// ---------------------------------------------------------------------------
// Per-type linear: out[n,:] = act(in[n,:] @ W[type[n]] + bias[type[n]])
// 256 threads = 8 waves; 16 rows x 128 cols per block; wave w owns cols
// [16w,16w+16). A staged in LDS once; B fragments streamed from L2 (f16,
// fragment-major). Computes all TTYPES variants, selects per-row at epilogue.
// ---------------------------------------------------------------------------
template<int K, int ACT>
__global__ __launch_bounds__(256) void pt_gemm(
    const float* __restrict__ in, const _Float16* __restrict__ Wf,
    const float* __restrict__ bias, const int* __restrict__ types,
    float* __restrict__ out, int n)
{
  constexpr int NK  = K / 32;
  constexpr int LDA = K + 8;
  __shared__ __align__(16) _Float16 sA[16 * LDA];

  const int row0 = blockIdx.x * 16;
  const int tid  = threadIdx.x;

  // Stage A tile (16 x K) as f16 into LDS: float4 loads, 8B LDS stores
  for (int idx = tid; idx < 16 * (K / 4); idx += 256) {
    int r  = idx / (K / 4);
    int k4 = idx - r * (K / 4);
    int gr = row0 + r;
    float4 v = make_float4(0.f, 0.f, 0.f, 0.f);
    if (gr < n) v = ((const float4*)(in + (size_t)gr * K))[k4];
    v4h h;
    h[0] = (_Float16)v.x; h[1] = (_Float16)v.y;
    h[2] = (_Float16)v.z; h[3] = (_Float16)v.w;
    *(v4h*)(sA + r * LDA + k4 * 4) = h;
  }
  __syncthreads();

  const int wave = tid >> 5;
  const int lane = tid & 31;

  // Hoist A fragments (shared across all 3 type-GEMMs)
  v16h afrag[NK];
#pragma unroll
  for (int kc = 0; kc < NK; ++kc) afrag[kc] = ldsA_frag(sA, LDA, kc);

  v8f acc0 = {}, acc1 = {}, acc2 = {};
#pragma unroll
  for (int t = 0; t < TTYPES; ++t) {
    v8f acc = {};
#pragma unroll
    for (int kc = 0; kc < NK; ++kc) {
      const _Float16* bp =
          Wf + ((((size_t)t * NK + kc) * 8 + wave) * 32 + lane) * 16;
      v16h b = *(const v16h*)bp;   // 32B contiguous per lane, L2-resident
      acc = __builtin_amdgcn_wmma_f32_16x16x32_f16(
          false, afrag[kc], false, b, (short)0, acc, false, false);
    }
    if (t == 0) acc0 = acc; else if (t == 1) acc1 = acc; else acc2 = acc;
  }

  // Epilogue: C layout — lane L, elem r -> row = r + (L>>4)*8, col = c0+(L&15)
  const int colL = wave * 16 + (lane & 15);
  const int mofs = (lane >> 4) * 8;
#pragma unroll
  for (int r = 0; r < 8; ++r) {
    int gr = row0 + r + mofs;
    if (gr >= n) continue;
    int t = types[gr];
    float v;
    if (t == 0)      v = acc0[r];
    else if (t == 1) v = acc1[r];
    else             v = acc2[r];
    v += bias[t * NHID + colL];
    if (ACT == 1) v = tanhf(v);
    out[(size_t)gr * NHID + colL] = v;
  }
}

// ---------------------------------------------------------------------------
// RTE precompute: rte = tab @ rte_w + rte_b  (240x256 @ 256x128)
// ---------------------------------------------------------------------------
__global__ void rte_proj(const float* __restrict__ tab, const float* __restrict__ w,
                         const float* __restrict__ b, float* __restrict__ rte)
{
  int m = blockIdx.x, o = threadIdx.x;
  float s = b[o];
  for (int d = 0; d < RTED; ++d) s += tab[m * RTED + d] * w[d * NHID + o];
  rte[m * NHID + o] = s;
}

// rtek[t,m,:] = rte[m,:] @ k_w[t];  rtev likewise (weight only, bias in k_node)
__global__ void rte_kv(const float* __restrict__ rte, const float* __restrict__ kw,
                       const float* __restrict__ vw, float* __restrict__ rtek,
                       float* __restrict__ rtev)
{
  int mt = blockIdx.x;
  int t = mt / MAXLEN, m = mt - t * MAXLEN;
  int o = threadIdx.x;
  const float* kwt = kw + (size_t)t * NHID * NHID;
  const float* vwt = vw + (size_t)t * NHID * NHID;
  float sk = 0.f, sv = 0.f;
  for (int d = 0; d < NHID; ++d) {
    float x = rte[m * NHID + d];
    sk += x * kwt[d * NHID + o];
    sv += x * vwt[d * NHID + o];
  }
  rtek[((size_t)t * MAXLEN + m) * NHID + o] = sk;
  rtev[((size_t)t * MAXLEN + m) * NHID + o] = sv;
}

// ---------------------------------------------------------------------------
__global__ void init_seg(float* __restrict__ aggr, float* __restrict__ nmax,
                         float* __restrict__ nsum, int n)
{
  int i = blockIdx.x * blockDim.x + threadIdx.x;
  if (i < n * NHID) aggr[i] = 0.0f;
  if (i < n * HEADS) { nmax[i] = -3.0e38f; nsum[i] = 0.0f; }
}

__device__ __forceinline__ void atomicMaxF(float* addr, float val) {
  unsigned int* ua = (unsigned int*)addr;
  unsigned int old = *ua;
  while (true) {
    float cur = __uint_as_float(old);
    if (cur >= val) break;
    unsigned int assumed = old;
    old = atomicCAS(ua, assumed, __float_as_uint(val));
    if (old == assumed) break;
  }
}

// ---------------------------------------------------------------------------
// Edge attention score: att[e,h] = <q_dst, ratt[r,h].(k_src + rte_k)> * pri/4
// one thread per (edge, head); all 16-float rows are 64B-aligned -> float4
// ---------------------------------------------------------------------------
__global__ void edge_score(const float* __restrict__ kbuf, const float* __restrict__ qbuf,
                           const float* __restrict__ rtek,
                           const int* __restrict__ src, const int* __restrict__ dst,
                           const int* __restrict__ etype, const int* __restrict__ etime,
                           const int* __restrict__ ntype,
                           const float* __restrict__ ratt, const float* __restrict__ pri,
                           float* __restrict__ att, float* __restrict__ nmax, int E)
{
  int gid = blockIdx.x * blockDim.x + threadIdx.x;
  if (gid >= E * HEADS) return;
  int e = gid >> 3, h = gid & 7;
  int s = src[e], d = dst[e], r = etype[e], tm = etime[e];
  int st = ntype[s];

  const float* kp = kbuf + (size_t)s * NHID + h * DKD;
  const float* rp = rtek + ((size_t)st * MAXLEN + tm) * NHID + h * DKD;
  float ke[DKD];
#pragma unroll
  for (int i = 0; i < 4; ++i) {
    float4 a = ld4(kp + 4 * i), b = ld4(rp + 4 * i);
    ke[4*i+0] = a.x + b.x; ke[4*i+1] = a.y + b.y;
    ke[4*i+2] = a.z + b.z; ke[4*i+3] = a.w + b.w;
  }

  const float* Wr = ratt + ((size_t)(r * HEADS + h)) * DKD * DKD;
  float k2[DKD];
#pragma unroll
  for (int j = 0; j < DKD; ++j) k2[j] = 0.0f;
  for (int dd = 0; dd < DKD; ++dd) {
    float kv = ke[dd];
    const float* wrow = Wr + dd * DKD;
#pragma unroll
    for (int i = 0; i < 4; ++i) {
      float4 w = ld4(wrow + 4 * i);
      k2[4*i+0] += kv * w.x; k2[4*i+1] += kv * w.y;
      k2[4*i+2] += kv * w.z; k2[4*i+3] += kv * w.w;
    }
  }

  const float* qp = qbuf + (size_t)d * NHID + h * DKD;
  float a = 0.0f;
#pragma unroll
  for (int i = 0; i < 4; ++i) {
    float4 q = ld4(qp + 4 * i);
    a += q.x * k2[4*i+0] + q.y * k2[4*i+1] + q.z * k2[4*i+2] + q.w * k2[4*i+3];
  }
  a *= pri[r * HEADS + h] * 0.25f;   // 1/sqrt(DK) = 0.25

  att[(size_t)e * HEADS + h] = a;
  atomicMaxF(&nmax[(size_t)d * HEADS + h], a);
}

__global__ void edge_exp(const float* __restrict__ nmax, const int* __restrict__ dst,
                         float* __restrict__ att, float* __restrict__ nsum, int E)
{
  int gid = blockIdx.x * blockDim.x + threadIdx.x;
  if (gid >= E * HEADS) return;
  int e = gid >> 3, h = gid & 7;
  int d = dst[e];
  float ex = __expf(att[gid] - nmax[(size_t)d * HEADS + h]);
  att[gid] = ex;
  atomicAdd(&nsum[(size_t)d * HEADS + h], ex);
}

// msg = (rmsg[r,h].(v_src + rte_v)) * softmax_w; atomic scatter into aggr[dst]
__global__ void edge_aggr(const float* __restrict__ vbuf, const float* __restrict__ rtev,
                          const int* __restrict__ src, const int* __restrict__ dst,
                          const int* __restrict__ etype, const int* __restrict__ etime,
                          const int* __restrict__ ntype,
                          const float* __restrict__ rmsg,
                          const float* __restrict__ att, const float* __restrict__ nsum,
                          float* __restrict__ aggr, int E)
{
  int gid = blockIdx.x * blockDim.x + threadIdx.x;
  if (gid >= E * HEADS) return;
  int e = gid >> 3, h = gid & 7;
  int s = src[e], d = dst[e], r = etype[e], tm = etime[e];
  int st = ntype[s];

  float w = att[gid] / (nsum[(size_t)d * HEADS + h] + 1e-16f);

  const float* vp = vbuf + (size_t)s * NHID + h * DKD;
  const float* rp = rtev + ((size_t)st * MAXLEN + tm) * NHID + h * DKD;
  float ve[DKD];
#pragma unroll
  for (int i = 0; i < 4; ++i) {
    float4 a = ld4(vp + 4 * i), b = ld4(rp + 4 * i);
    ve[4*i+0] = a.x + b.x; ve[4*i+1] = a.y + b.y;
    ve[4*i+2] = a.z + b.z; ve[4*i+3] = a.w + b.w;
  }

  const float* Wr = rmsg + ((size_t)(r * HEADS + h)) * DKD * DKD;
  float v2[DKD];
#pragma unroll
  for (int j = 0; j < DKD; ++j) v2[j] = 0.0f;
  for (int dd = 0; dd < DKD; ++dd) {
    float vv = ve[dd];
    const float* wrow = Wr + dd * DKD;
#pragma unroll
    for (int i = 0; i < 4; ++i) {
      float4 ww = ld4(wrow + 4 * i);
      v2[4*i+0] += vv * ww.x; v2[4*i+1] += vv * ww.y;
      v2[4*i+2] += vv * ww.z; v2[4*i+3] += vv * ww.w;
    }
  }

  float* ap = aggr + (size_t)d * NHID + h * DKD;
#pragma unroll
  for (int j = 0; j < DKD; ++j) atomicAdd(&ap[j], v2[j] * w);
}

// exact gelu (approximate=False): 0.5*x*(1+erf(x/sqrt(2))), in place
__global__ void gelu_k(float* __restrict__ x, int n)
{
  int i = blockIdx.x * blockDim.x + threadIdx.x;
  if (i >= n * NHID) return;
  float v = x[i];
  x[i] = 0.5f * v * (1.0f + erff(v * 0.70710678118654752f));
}

// out = trans*sigmoid(skip[type]) + x*(1-sigmoid(skip[type]))
__global__ void blend_k(const float* __restrict__ trans, const float* __restrict__ xin,
                        const float* __restrict__ skip, const int* __restrict__ types,
                        float* __restrict__ out, int n)
{
  int i = blockIdx.x * blockDim.x + threadIdx.x;
  if (i >= n * NHID) return;
  int node = i >> 7;   // /NHID
  int t = types[node];
  float a = 1.0f / (1.0f + __expf(-skip[t]));
  out[i] = trans[i] * a + xin[i] * (1.0f - a);
}

// ---------------------------------------------------------------------------
extern "C" void kernel_launch(void* const* d_in, const int* in_sizes, int n_in,
                              void* d_out, int out_size, void* d_ws, size_t ws_size,
                              hipStream_t stream)
{
  const float* node_feature = (const float*)d_in[0];
  const float* adapt_w = (const float*)d_in[1];
  const float* adapt_b = (const float*)d_in[2];
  const float* k_w = (const float*)d_in[3];
  const float* k_b = (const float*)d_in[4];
  const float* q_w = (const float*)d_in[5];
  const float* q_b = (const float*)d_in[6];
  const float* v_w = (const float*)d_in[7];
  const float* v_b = (const float*)d_in[8];
  const float* a_w = (const float*)d_in[9];
  const float* a_b = (const float*)d_in[10];
  const float* rel_pri = (const float*)d_in[11];
  const float* rel_att = (const float*)d_in[12];
  const float* rel_msg = (const float*)d_in[13];
  const float* skip = (const float*)d_in[14];
  const float* rte_tab = (const float*)d_in[15];
  const float* rte_w = (const float*)d_in[16];
  const float* rte_b = (const float*)d_in[17];
  const int* node_type  = (const int*)d_in[18];
  const int* edge_index = (const int*)d_in[19];
  const int* edge_type  = (const int*)d_in[20];
  const int* edge_time  = (const int*)d_in[21];

  const int N   = in_sizes[18];
  const int E   = in_sizes[20];
  const int IND = in_sizes[0] / N;           // 256
  const int* src = edge_index;
  const int* dst = edge_index + E;

  float* ws = (float*)d_ws;
  size_t o = 0;
  float* xbuf = ws + o; o += (size_t)N * NHID;
  float* kbuf = ws + o; o += (size_t)N * NHID;   // reused as `trans`
  float* qbuf = ws + o; o += (size_t)N * NHID;
  float* vbuf = ws + o; o += (size_t)N * NHID;
  float* aggr = ws + o; o += (size_t)N * NHID;
  float* att  = ws + o; o += (size_t)E * HEADS;
  float* nmax = ws + o; o += (size_t)N * HEADS;
  float* nsum = ws + o; o += (size_t)N * HEADS;
  float* rteb = ws + o; o += (size_t)MAXLEN * NHID;
  float* rtek = ws + o; o += (size_t)TTYPES * MAXLEN * NHID;
  float* rtev = ws + o; o += (size_t)TTYPES * MAXLEN * NHID;
  o = (o + 15) & ~(size_t)15;   // 64B-align the f16 region

  // f16 fragment-major weight copies
  _Float16* hws = (_Float16*)(ws + o);
  size_t ho = 0;
  _Float16* wf_adapt = hws + ho; ho += (size_t)TTYPES * IND * NHID;
  _Float16* wf_k     = hws + ho; ho += (size_t)2 * TTYPES * NHID * NHID;
  _Float16* wf_q     = hws + ho; ho += (size_t)2 * TTYPES * NHID * NHID;
  _Float16* wf_v     = hws + ho; ho += (size_t)2 * TTYPES * NHID * NHID;
  _Float16* wf_a     = hws + ho; ho += (size_t)2 * TTYPES * NHID * NHID;
  (void)ws_size;

  const int ntile     = (N + 15) / 16;
  const int eh_blocks = (E * HEADS + 255) / 256;
  const int nc_blocks = (N * NHID + 255) / 256;

  // --- Weight conversion (tiny, L2-resident afterwards) ---
  {
    int tot_a = TTYPES * IND * NHID;
    convert_w<<<(tot_a + 255) / 256, 256, 0, stream>>>(adapt_w, wf_adapt, IND, tot_a);
    int tot = 2 * TTYPES * NHID * NHID;   // both layers: t-major works layer-blocked
    convert_w<<<(tot + 255) / 256, 256, 0, stream>>>(k_w, wf_k, NHID, tot);
    convert_w<<<(tot + 255) / 256, 256, 0, stream>>>(q_w, wf_q, NHID, tot);
    convert_w<<<(tot + 255) / 256, 256, 0, stream>>>(v_w, wf_v, NHID, tot);
    convert_w<<<(tot + 255) / 256, 256, 0, stream>>>(a_w, wf_a, NHID, tot);
  }

  // adapt: x = tanh(per_type_linear(node_feature))
  pt_gemm<256, 1><<<ntile, 256, 0, stream>>>(node_feature, wf_adapt, adapt_b,
                                             node_type, xbuf, N);

  const size_t lwf = (size_t)TTYPES * NHID * NHID;   // f16 elems per layer block
  for (int l = 0; l < 2; ++l) {
    const float* kw = k_w + (size_t)l * TTYPES * NHID * NHID;
    const float* vw = v_w + (size_t)l * TTYPES * NHID * NHID;
    const float* kb = k_b + (size_t)l * TTYPES * NHID;
    const float* qb = q_b + (size_t)l * TTYPES * NHID;
    const float* vb = v_b + (size_t)l * TTYPES * NHID;
    const float* ab = a_b + (size_t)l * TTYPES * NHID;
    const float* pri  = rel_pri + (size_t)l * RREL * HEADS;
    const float* ratt = rel_att + (size_t)l * RREL * HEADS * DKD * DKD;
    const float* rmsg = rel_msg + (size_t)l * RREL * HEADS * DKD * DKD;

    // RTE tables (tiny)
    rte_proj<<<MAXLEN, NHID, 0, stream>>>(rte_tab + (size_t)l * MAXLEN * RTED,
                                          rte_w + (size_t)l * RTED * NHID,
                                          rte_b + (size_t)l * NHID, rteb);
    rte_kv<<<TTYPES * MAXLEN, NHID, 0, stream>>>(rteb, kw, vw, rtek, rtev);

    // k/q/v projections (WMMA, B-fragments from L2)
    pt_gemm<128, 0><<<ntile, 256, 0, stream>>>(xbuf, wf_k + l * lwf, kb, node_type, kbuf, N);
    pt_gemm<128, 0><<<ntile, 256, 0, stream>>>(xbuf, wf_q + l * lwf, qb, node_type, qbuf, N);
    pt_gemm<128, 0><<<ntile, 256, 0, stream>>>(xbuf, wf_v + l * lwf, vb, node_type, vbuf, N);

    // segment softmax + aggregation
    init_seg<<<nc_blocks, 256, 0, stream>>>(aggr, nmax, nsum, N);
    edge_score<<<eh_blocks, 256, 0, stream>>>(kbuf, qbuf, rtek, src, dst, edge_type,
                                              edge_time, node_type, ratt, pri,
                                              att, nmax, E);
    edge_exp<<<eh_blocks, 256, 0, stream>>>(nmax, dst, att, nsum, E);
    edge_aggr<<<eh_blocks, 256, 0, stream>>>(vbuf, rtev, src, dst, edge_type,
                                             edge_time, node_type, rmsg, att, nsum,
                                             aggr, E);

    // update: gelu -> per-type linear (WMMA) -> gated blend
    gelu_k<<<nc_blocks, 256, 0, stream>>>(aggr, N);
    pt_gemm<128, 0><<<ntile, 256, 0, stream>>>(aggr, wf_a + l * lwf, ab, node_type, kbuf, N);
    blend_k<<<nc_blocks, 256, 0, stream>>>(kbuf, xbuf, skip + (size_t)l * TTYPES,
                                           node_type,
                                           (l == 0) ? xbuf : (float*)d_out, N);
  }
}
